// CustomTwoLayerGNN_34333968564342
// MI455X (gfx1250) — matrix operand order repaired and checked by reading, not compiled, not run on hardware
//
#include <hip/hip_runtime.h>
#include <hip/hip_bf16.h>

// GNN dims (fixed by the reference)
#define IN_FEATS  128
#define HID_FEATS 256
#define OUT_FEATS 256

typedef __bf16 bf16_t;
typedef __attribute__((ext_vector_type(16))) __bf16 v16bf;
typedef __attribute__((ext_vector_type(8)))  float  v8f;

__device__ __forceinline__ void atom_add_f32(float* p, float v) {
    __hip_atomic_fetch_add(p, v, __ATOMIC_RELAXED, __HIP_MEMORY_SCOPE_AGENT);
}

__device__ __forceinline__ float bfbits_to_f(unsigned short u) {
    unsigned int i = ((unsigned int)u) << 16;
    return __builtin_bit_cast(float, i);
}

__device__ __forceinline__ void store_out(float* p, float v)  { *p = v; }
__device__ __forceinline__ void store_out(bf16_t* p, float v) { *p = (__bf16)v; }

// ---------------------------------------------------------------------------
// deg -> inv_deg in place (IEEE divide once per node, outside the GEMM)
// ---------------------------------------------------------------------------
__global__ void invdeg_kernel(float* __restrict__ deg, int M) {
    int i = blockIdx.x * blockDim.x + threadIdx.x;
    if (i < M) {
        float d = deg[i];
        deg[i] = d > 0.0f ? 1.0f / d : 0.0f;
    }
}

// ---------------------------------------------------------------------------
// Bulk f32 -> bf16 conversion (x -> xbf), 4 elems/thread, coalesced
// ---------------------------------------------------------------------------
__global__ void cvt_bf16_kernel(const float* __restrict__ in, bf16_t* __restrict__ out,
                                long long n4) {
    long long t = (long long)blockIdx.x * blockDim.x + threadIdx.x;
    if (t >= n4) return;
    float4 v = ((const float4*)in)[t];
    bf16_t* o = out + t * 4;
    o[0] = (__bf16)v.x; o[1] = (__bf16)v.y; o[2] = (__bf16)v.z; o[3] = (__bf16)v.w;
}

// ---------------------------------------------------------------------------
// Fused normalize + convert: out_bf16[row, :] = agg_f32[row, :] * inv_deg[row]
// ---------------------------------------------------------------------------
template<int FEATS>
__global__ void norm_bf16_kernel(const float* __restrict__ agg,
                                 const float* __restrict__ invdeg,
                                 bf16_t* __restrict__ out, int M) {
    const int CHUNKS = FEATS / 4;
    long long t = (long long)blockIdx.x * blockDim.x + threadIdx.x;
    int row = (int)(t / CHUNKS);
    int c   = (int)(t % CHUNKS);
    if (row >= M) return;
    float s = invdeg[row];
    float4 v = ((const float4*)(agg + (size_t)row * FEATS))[c];
    bf16_t* o = out + (size_t)row * FEATS + (size_t)c * 4;
    o[0] = (__bf16)(v.x * s); o[1] = (__bf16)(v.y * s);
    o[2] = (__bf16)(v.z * s); o[3] = (__bf16)(v.w * s);
}

// ---------------------------------------------------------------------------
// Layer-1 scatter: agg1[dst[e]] += x[src[e]] (128 f32), fused degree count.
// ---------------------------------------------------------------------------
template<int FEATS>
__global__ void scatter_f32_deg(const float* __restrict__ h, const int* __restrict__ src,
                                const int* __restrict__ dstv, float* __restrict__ agg,
                                float* __restrict__ deg, int E) {
    const int CHUNKS = FEATS / 4;
    long long tid = (long long)blockIdx.x * blockDim.x + threadIdx.x;
    int e = (int)(tid / CHUNKS);
    int c = (int)(tid % CHUNKS);
    if (e >= E) return;
    int s = src[e], d = dstv[e];
    if (c == 0) atom_add_f32(&deg[d], 1.0f);     // fused in-degree
    float4 v = ((const float4*)(h + (size_t)s * FEATS))[c];
    float* o = agg + (size_t)d * FEATS + (size_t)c * 4;
    atom_add_f32(o + 0, v.x);
    atom_add_f32(o + 1, v.y);
    atom_add_f32(o + 2, v.z);
    atom_add_f32(o + 3, v.w);
}

// Layer-2 scatter: source rows are bf16 (h1) -> halves gather traffic.
template<int FEATS>
__global__ void scatter_bf16(const unsigned short* __restrict__ h, const int* __restrict__ src,
                             const int* __restrict__ dstv, float* __restrict__ agg, int E) {
    const int CHUNKS = FEATS / 4;
    long long tid = (long long)blockIdx.x * blockDim.x + threadIdx.x;
    int e = (int)(tid / CHUNKS);
    int c = (int)(tid % CHUNKS);
    if (e >= E) return;
    int s = src[e], d = dstv[e];
    ushort4 v = ((const ushort4*)(h + (size_t)s * FEATS))[c];
    float* o = agg + (size_t)d * FEATS + (size_t)c * 4;
    atom_add_f32(o + 0, bfbits_to_f(v.x));
    atom_add_f32(o + 1, bfbits_to_f(v.y));
    atom_add_f32(o + 2, bfbits_to_f(v.z));
    atom_add_f32(o + 3, bfbits_to_f(v.w));
}

// ---------------------------------------------------------------------------
// Fused SAGE layer GEMM:  out = relu( concat(A0bf, A1bf) @ W + bias )
// A0bf/A1bf pre-converted bf16 [M, KHALF] (A1bf already inv_deg-scaled).
//
// Wave32 WMMA (v_wmma_f32_16x16x32_bf16), fragment layouts used here (contiguous k order, same for A and B):
//   A (16x32 bf16): lane row = tile_m*16 + (lane&15),
//                   K = kk + (lane>>4)*16 + j  -> per lane one
//                   contiguous 32B run => two global_load_b128, NO cvts
//   B (32x16 bf16): lane col = tile_n*16 + (lane&15), K = kk + (lane>>4)*16 + j
//   C/D (16x16 f32): VGPR r -> row (lane>>4)*8 + r, col = lane&15
//
// B fragments for the FULL K are register-resident, amortized over MITER
// m-tiles. A fragments are even/odd double-buffered, and sched_group_barrier
// pins the schedule to  VMEM*4, (WMMA, VMEM*2)*(NB-1), WMMA  so one fragment's
// loads are always in flight behind the WMMA consuming the previous fragment.
// ---------------------------------------------------------------------------
template<int KHALF, int MITER, typename TOut>
__global__ __launch_bounds__(128)
void gemm_concat_bias_relu(const bf16_t* __restrict__ A0,
                           const bf16_t* __restrict__ A1,
                           const float* __restrict__ W,
                           const float* __restrict__ bias,
                           TOut* __restrict__ outp,
                           int Mtiles, int N) {
    const int NB  = (2 * KHALF) / 32;    // fragments over full K (8 or 16, even)
    const int NBH = NB / 2;              // fragments per K-half

    int wave = threadIdx.x >> 5;         // 4 waves per block
    int lane = threadIdx.x & 31;
    int tile_n = blockIdx.x;
    int m0 = blockIdx.y * (4 * MITER);   // first m-tile of this block

    int half = lane >> 4;                // 0 or 1
    int l16  = lane & 15;
    int col  = tile_n * 16 + l16;        // B/C/D column for this lane

    // ---- preload ALL B fragments for full K into registers ----
    v16bf bfrag[NB];
#pragma unroll
    for (int f = 0; f < NB; ++f) {
        int kk = f * 32;
#pragma unroll
        for (int j = 0; j < 16; ++j) {
            int k = kk + half * 16 + j;
            bfrag[f][j] = (__bf16)W[(size_t)k * N + col];
        }
    }

    float bv = bias[col];

    // ---- iterate m-tiles, re-using the register-resident B ----
    for (int i = 0; i < MITER; ++i) {
        int tm = m0 + wave + 4 * i;      // wave-uniform
        if (tm >= Mtiles) break;         // uniform branch: EXEC stays all-1s
        int rowoff = (tm * 16 + l16) * KHALF;   // fits in 32 bits

        const bf16_t* a0 = A0 + rowoff;
        const bf16_t* a1 = A1 + rowoff;

        // unified fragment loader over both K-halves (f const after unroll)
        auto loadfrag = [&](v16bf& a, int f) {
            const bf16_t* base = (f < NBH) ? a0 : a1;
            int kk = (f < NBH ? f : f - NBH) * 32;
#pragma unroll
            for (int j = 0; j < 16; ++j) {
                int k = kk + half * 16 + j;
                a[j] = base[k];
            }
        };

        v8f acc = {};
        v16bf aeven, aodd;               // two live buffers -> pipelined loads

        loadfrag(aeven, 0);
#pragma unroll
        for (int f = 0; f < NB; f += 2) {
            loadfrag(aodd, f + 1);       // in flight during WMMA f
            acc = __builtin_amdgcn_wmma_f32_16x16x32_bf16(
                false, aeven, false, bfrag[f], (short)0, acc, false, false);
            if (f + 2 < NB) loadfrag(aeven, f + 2);   // in flight during WMMA f+1
            acc = __builtin_amdgcn_wmma_f32_16x16x32_bf16(
                false, aodd, false, bfrag[f + 1], (short)0, acc, false, false);
        }

        // ---- scheduler guidance: keep one load-pair ahead of each WMMA ----
        // 0x020 = VMEM read group, 0x008 = MFMA/WMMA group
        __builtin_amdgcn_sched_group_barrier(0x020, 4, 0);   // pairs f=0, f=1
#pragma unroll
        for (int f = 0; f < NB - 1; ++f) {
            __builtin_amdgcn_sched_group_barrier(0x008, 1, 0); // WMMA f
            __builtin_amdgcn_sched_group_barrier(0x020, 2, 0); // pair f+2
        }
        __builtin_amdgcn_sched_group_barrier(0x008, 1, 0);     // last WMMA

        // epilogue: bias + ReLU, store per C/D layout (32-bit addressing)
        int obase = (tm * 16 + half * 8) * N + col;
#pragma unroll
        for (int r = 0; r < 8; ++r) {
            float v = acc[r] + bv;
            v = v > 0.0f ? v : 0.0f;
            store_out(&outp[obase + r * N], v);
        }
    }
}

// ---------------------------------------------------------------------------
extern "C" void kernel_launch(void* const* d_in, const int* in_sizes, int n_in,
                              void* d_out, int out_size, void* d_ws, size_t ws_size,
                              hipStream_t stream) {
    const float* x  = (const float*)d_in[0];
    const int*   src  = (const int*)d_in[1];
    const int*   dstv = (const int*)d_in[2];
    const float* W1 = (const float*)d_in[3];
    const float* b1 = (const float*)d_in[4];
    const float* W2 = (const float*)d_in[5];
    const float* b2 = (const float*)d_in[6];
    float* out = (float*)d_out;

    const int M = in_sizes[0] / IN_FEATS;   // 50000
    const int E = in_sizes[1];              // 800000
    const int Mtiles = (M + 15) / 16;       // 3125 (M divisible by 16)
    const int MITER = 8;                    // m-tiles per wave in GEMM

    // Workspace layout (accumulators contiguous so one memset zeroes them):
    //  deg[M]f32 | agg1[M,128]f32 | agg2[M,256]f32 | h1[M,256]bf16 |
    //  xbf[M,128]bf16 | a1bf[M,128]bf16 | a2bf[M,256]bf16
    char* ws = (char*)d_ws;
    size_t o_deg  = 0;
    size_t o_agg1 = (o_deg  + (size_t)M * 4             + 255) & ~(size_t)255;
    size_t o_agg2 = (o_agg1 + (size_t)M * IN_FEATS  * 4 + 255) & ~(size_t)255;
    size_t o_h1   = (o_agg2 + (size_t)M * HID_FEATS * 4 + 255) & ~(size_t)255;
    size_t o_xbf  = (o_h1   + (size_t)M * HID_FEATS * 2 + 255) & ~(size_t)255;
    size_t o_a1bf = (o_xbf  + (size_t)M * IN_FEATS  * 2 + 255) & ~(size_t)255;
    size_t o_a2bf = (o_a1bf + (size_t)M * IN_FEATS  * 2 + 255) & ~(size_t)255;
    float*  deg  = (float*)(ws + o_deg);
    float*  agg1 = (float*)(ws + o_agg1);
    float*  agg2 = (float*)(ws + o_agg2);
    bf16_t* h1   = (bf16_t*)(ws + o_h1);
    bf16_t* xbf  = (bf16_t*)(ws + o_xbf);
    bf16_t* a1bf = (bf16_t*)(ws + o_a1bf);
    bf16_t* a2bf = (bf16_t*)(ws + o_a2bf);

    // Zero deg + agg1 + agg2 every call (graph-capturable async memset)
    (void)hipMemsetAsync(ws, 0, o_h1, stream);

    // 0) x (f32) -> xbf (bf16), once
    long long n4 = (long long)M * IN_FEATS / 4;
    cvt_bf16_kernel<<<(unsigned)((n4 + 255) / 256), 256, 0, stream>>>(x, xbf, n4);

    // 1) layer-1 scatter-sum of x[src] into agg1[dst], fused in-degree count
    long long w1 = (long long)E * (IN_FEATS / 4);
    scatter_f32_deg<IN_FEATS><<<(unsigned)((w1 + 255) / 256), 256, 0, stream>>>(
        x, src, dstv, agg1, deg, E);

    // 2) deg -> inv_deg (in place)
    invdeg_kernel<<<(M + 255) / 256, 256, 0, stream>>>(deg, M);

    // 3) a1bf = bf16(agg1 * inv_deg)
    long long t1 = (long long)M * (IN_FEATS / 4);
    norm_bf16_kernel<IN_FEATS><<<(unsigned)((t1 + 255) / 256), 256, 0, stream>>>(
        agg1, deg, a1bf, M);

    // 4) h1 = relu(concat(x, mean) @ W1 + b1), stored bf16
    dim3 g1(HID_FEATS / 16, (Mtiles + 4 * MITER - 1) / (4 * MITER));
    gemm_concat_bias_relu<IN_FEATS, 8, bf16_t><<<g1, 128, 0, stream>>>(
        xbf, a1bf, W1, b1, h1, Mtiles, HID_FEATS);

    // 5) layer-2 scatter-sum of h1[src] (bf16) into agg2[dst] (f32)
    long long w2 = (long long)E * (HID_FEATS / 4);
    scatter_bf16<HID_FEATS><<<(unsigned)((w2 + 255) / 256), 256, 0, stream>>>(
        (const unsigned short*)h1, src, dstv, agg2, E);

    // 6) a2bf = bf16(agg2 * inv_deg)
    long long t2 = (long long)M * (HID_FEATS / 4);
    norm_bf16_kernel<HID_FEATS><<<(unsigned)((t2 + 255) / 256), 256, 0, stream>>>(
        agg2, deg, a2bf, M);

    // 7) out = relu(concat(h1, mean) @ W2 + b2), fp32
    dim3 g2(OUT_FEATS / 16, (Mtiles + 4 * MITER - 1) / (4 * MITER));
    gemm_concat_bias_relu<HID_FEATS, 8, float><<<g2, 128, 0, stream>>>(
        h1, a2bf, W2, b2, out, Mtiles, OUT_FEATS);

    (void)n_in; (void)out_size; (void)ws_size;
}
